// LoRA_80839874445852
// MI455X (gfx1250) — matrix operand order repaired and checked by reading, not compiled, not run on hardware
//
#include <hip/hip_runtime.h>

// ---------------- CDNA5 types ----------------
typedef __attribute__((ext_vector_type(16))) __bf16 v16bf;
typedef __attribute__((ext_vector_type(2)))  __bf16 v2bf;
typedef __attribute__((ext_vector_type(8)))  float  v8f;
typedef __attribute__((ext_vector_type(4)))  unsigned int v4u;
typedef __attribute__((ext_vector_type(8)))  int v8i;
typedef __attribute__((ext_vector_type(4)))  int v4i;

union AFrag { v16bf v; uint4 q[2]; unsigned int u[8]; };
union CFrag { v8f v; float f[8]; };
union P2 { v2bf b; unsigned int u; };
union P1 { __bf16 b; unsigned short u; };

// hardware packed f32->bf16 (v_cvt_pk_bf16_f32) via cast
__device__ __forceinline__ unsigned int cvt2(float lo, float hi) {
  P2 p; p.b[0] = (__bf16)lo; p.b[1] = (__bf16)hi; return p.u;
}
__device__ __forceinline__ unsigned short cvt1(float f) {
  P1 p; p.b = (__bf16)f; return p.u;
}

// Problem constants (from reference setup)
#define BS   8
#define SQ   2048
#define D    4096
#define M_TOT (BS * SQ)     // 16384
#define RK   16

// ---------------- TDM: issue one 2D tile load (128 rows x 64 bf16) ----------------
// D# per CDNA5 ISA. Row stride in memory = 4096 elements,
// LDS dest padded 4 DWORDs after every 32 DWORDs (row = 128B) -> 144B LDS row stride.
__device__ __forceinline__ void tdm_load_w_tile(const unsigned short* gptr,
                                                unsigned int ldsAddr) {
  unsigned long long ga = (unsigned long long)(const void*)gptr;
  v4u g0;
  g0[0] = 1u;                                    // count=1, user mode, no gather
  g0[1] = ldsAddr;                               // lds_addr (bytes)
  g0[2] = (unsigned int)(ga & 0xFFFFFFFFu);      // global_addr[31:0]
  g0[3] = (unsigned int)((ga >> 32) & 0x01FFFFFFu) | (2u << 30); // addr[56:32] | type=2
  v8i g1;
  g1[0] = (int)((1u << 16)      // data_size = 2 bytes
              | (1u << 20)      // pad_enable
              | (4u << 22)      // pad_interval: after every 32 DWORDs
              | (3u << 25));    // pad_amount: 4 DWORDs (16B)
  g1[1] = (int)(4096u << 16);   // tensor_dim0[15:0] << 16 (dim0 = 4096 elems)
  g1[2] = (int)(4096u << 16);   // tensor_dim0[31:16]=0 | tensor_dim1 low16 = 4096
  g1[3] = (int)(64u << 16);     // tensor_dim1 hi=0 | tile_dim0 = 64
  g1[4] = 128;                  // tile_dim1 = 128 | tile_dim2 = 0
  g1[5] = 4096;                 // tensor_dim0_stride low32 = 4096 elems
  g1[6] = 0;                    // dim0_stride hi | dim1_stride low
  g1[7] = 0;
  v4i z; z[0] = 0; z[1] = 0; z[2] = 0; z[3] = 0;   // 2D: groups 2/3 unused
#if defined(__clang_major__) && (__clang_major__ >= 23)
  v8i z8; z8[0]=0; z8[1]=0; z8[2]=0; z8[3]=0; z8[4]=0; z8[5]=0; z8[6]=0; z8[7]=0;
  __builtin_amdgcn_tensor_load_to_lds(g0, g1, z, z, z8, 0);
#else
  __builtin_amdgcn_tensor_load_to_lds(g0, g1, z, z, 0);
#endif
}

// ---------------- conversion kernels ----------------
__global__ void convert_w_bf16(const float* __restrict__ W,
                               unsigned int* __restrict__ Wb, int n2) {
  int i = blockIdx.x * blockDim.x + threadIdx.x;
  if (i < n2) {
    float2 f = ((const float2*)W)[i];
    Wb[i] = cvt2(f.x, f.y);
  }
}

// A1[b][r][d] = A[b/2][d/1024][d%1024][r]  (column-major A_exp, bf16)
__global__ void build_A1(const float* __restrict__ A,
                         unsigned short* __restrict__ A1) {
  int idx = blockIdx.x * blockDim.x + threadIdx.x;   // over 8*16*4096
  if (idx >= BS * RK * D) return;
  int d = idx & (D - 1);
  int r = (idx >> 12) & (RK - 1);
  int b = idx >> 16;
  int split = b >> 1;
  int skill = d >> 10;
  int dd = d & 1023;
  A1[idx] = cvt1(A[(((split * 4 + skill) * 1024) + dd) * RK + r]);
}

// Beps[b][o][k] = (k<16) ? B[b/2][o/1024][k][o%1024] : 0   (K padded to 32)
__global__ void build_Beps(const float* __restrict__ B,
                           unsigned short* __restrict__ Beps) {
  int idx = blockIdx.x * blockDim.x + threadIdx.x;   // over 8*4096*32
  if (idx >= BS * D * 32) return;
  int k = idx & 31;
  int o = (idx >> 5) & (D - 1);
  int b = idx >> 17;
  int split = b >> 1, skill = o >> 10, col = o & 1023;
  float v = (k < RK) ? B[(((split * 4 + skill) * RK) + k) * 1024 + col] : 0.0f;
  Beps[idx] = cvt1(v);
}

// ---------------- stage 1: t2 = (x @ A_exp) / 16, bf16 [M_TOT][32] ----------------
__global__ __launch_bounds__(256)
void lora_stage1(const float* __restrict__ x,
                 const unsigned short* __restrict__ A1,
                 unsigned short* __restrict__ t2) {
  const int lane = threadIdx.x & 31;
  const int wave = threadIdx.x >> 5;
  const int hi = lane >> 4, l15 = lane & 15;
  const int m0 = blockIdx.x * 128 + wave * 16;
  const int batch = m0 >> 11;
  const float* xrow = x + (size_t)(m0 + l15) * D;
  const unsigned short* acol = A1 + ((size_t)batch * RK + l15) * D;

  CFrag acc;
  #pragma unroll
  for (int i = 0; i < 8; ++i) acc.f[i] = 0.0f;

  for (int k = 0; k < D; k += 32) {
    AFrag a, bfr;
    const int c0 = k + hi * 8;
    float4 f0 = *(const float4*)(xrow + c0);
    float4 f1 = *(const float4*)(xrow + c0 + 4);
    float4 f2 = *(const float4*)(xrow + c0 + 16);
    float4 f3 = *(const float4*)(xrow + c0 + 20);
    a.u[0] = cvt2(f0.x, f0.y); a.u[1] = cvt2(f0.z, f0.w);
    a.u[2] = cvt2(f1.x, f1.y); a.u[3] = cvt2(f1.z, f1.w);
    a.u[4] = cvt2(f2.x, f2.y); a.u[5] = cvt2(f2.z, f2.w);
    a.u[6] = cvt2(f3.x, f3.y); a.u[7] = cvt2(f3.z, f3.w);
    const unsigned short* bp = acol + k + hi * 8;
    bfr.q[0] = *(const uint4*)bp;
    bfr.q[1] = *(const uint4*)(bp + 16);
    acc.v = __builtin_amdgcn_wmma_f32_16x16x32_bf16(
        false, a.v, false, bfr.v, (short)0, acc.v, false, false);
  }

  #pragma unroll
  for (int vv = 0; vv < 8; ++vv) {
    int row = m0 + vv + 8 * hi;
    t2[(size_t)row * 32 + l15]      = cvt1(acc.f[vv] * 0.0625f);
    t2[(size_t)row * 32 + l15 + 16] = 0;   // zero-padded K half
  }
}

// ---------------- main GEMM: out = x @ Wt + bias + t2 @ Beps ----------------
// Double-buffered LDS; W tiles via Tensor Data Mover, X tiles via regs + cvt.
__global__ __launch_bounds__(256)
void lora_gemm(const float* __restrict__ x,
               const unsigned short* __restrict__ Wb,
               const float* __restrict__ bias,
               const unsigned short* __restrict__ t2,
               const unsigned short* __restrict__ Beps,
               float* __restrict__ out) {
  __shared__ __align__(16) unsigned short lX[2][128][72];
  __shared__ __align__(16) unsigned short lW[2][128][72];

  const int tid  = threadIdx.x;
  const int lane = tid & 31;
  const int wave = tid >> 5;
  const int wm = wave >> 2;      // 0..1  (64-row half)
  const int wn = wave & 3;       // 0..3  (32-col quarter)
  const int hi = lane >> 4;
  const int l15 = lane & 15;

  const int mBlk = blockIdx.y * 128;
  const int nBlk = blockIdx.x * 128;

  CFrag acc[4][2];
  #pragma unroll
  for (int i = 0; i < 4; ++i)
    #pragma unroll
    for (int j = 0; j < 2; ++j)
      #pragma unroll
      for (int v = 0; v < 8; ++v) acc[i][j].f[v] = 0.0f;

  float4 xr[8];
  // ---- prologue: stage kt=0 ----
  #pragma unroll
  for (int i = 0; i < 8; ++i) {
    int f = tid + i * 256, row = f >> 4, c4 = (f & 15) * 4;
    xr[i] = *(const float4*)(x + (size_t)(mBlk + row) * D + c4);
  }
  if (wave == 0)
    tdm_load_w_tile(Wb + (size_t)nBlk * D,
                    (unsigned int)(size_t)&lW[0][0][0]);
  #pragma unroll
  for (int i = 0; i < 8; ++i) {
    int f = tid + i * 256, row = f >> 4, c4 = (f & 15) * 4;
    uint2 p;
    p.x = cvt2(xr[i].x, xr[i].y);
    p.y = cvt2(xr[i].z, xr[i].w);
    *(uint2*)&lX[0][row][c4] = p;
  }
  if (wave == 0) __builtin_amdgcn_s_wait_tensorcnt(0);
  __syncthreads();

  for (int kt = 0; kt < D / 64; ++kt) {
    const int cur = kt & 1, nxt = cur ^ 1;
    const bool more = (kt + 1) < (D / 64);

    // issue next-stage data movement before compute (overlap)
    if (more) {
      const int k1 = (kt + 1) * 64;
      #pragma unroll
      for (int i = 0; i < 8; ++i) {
        int f = tid + i * 256, row = f >> 4, c4 = (f & 15) * 4;
        xr[i] = *(const float4*)(x + (size_t)(mBlk + row) * D + k1 + c4);
      }
      if (wave == 0)
        tdm_load_w_tile(Wb + (size_t)nBlk * D + k1,
                        (unsigned int)(size_t)&lW[nxt][0][0]);
      if (kt + 2 < D / 64)
        __builtin_prefetch(x + (size_t)(mBlk + (tid >> 1)) * D + k1 + 64, 0, 0);
    }

    // ---- compute current stage ----
    #pragma unroll
    for (int ks = 0; ks < 2; ++ks) {
      const int kk = ks * 32;
      AFrag a[4], bfr[2];
      const int cA = kk + hi * 8;
      #pragma unroll
      for (int i = 0; i < 4; ++i) {
        const unsigned short* p = &lX[cur][wm * 64 + i * 16 + l15][cA];
        a[i].q[0] = *(const uint4*)p;          // K c..c+7
        a[i].q[1] = *(const uint4*)(p + 16);   // K c+16..c+23
      }
      const int cB = kk + hi * 8;
      #pragma unroll
      for (int j = 0; j < 2; ++j) {
        const unsigned short* p = &lW[cur][wn * 32 + j * 16 + l15][cB];
        bfr[j].q[0] = *(const uint4*)p;        // K cB..cB+7
        bfr[j].q[1] = *(const uint4*)(p + 16); // K cB+16..cB+23
      }
      #pragma unroll
      for (int i = 0; i < 4; ++i)
        #pragma unroll
        for (int j = 0; j < 2; ++j)
          acc[i][j].v = __builtin_amdgcn_wmma_f32_16x16x32_bf16(
              false, a[i].v, false, bfr[j].v, (short)0, acc[i][j].v, false, false);
    }

    // commit next-stage X to LDS, wait TDM, sync
    if (more) {
      #pragma unroll
      for (int i = 0; i < 8; ++i) {
        int f = tid + i * 256, row = f >> 4, c4 = (f & 15) * 4;
        uint2 p;
        p.x = cvt2(xr[i].x, xr[i].y);
        p.y = cvt2(xr[i].z, xr[i].w);
        *(uint2*)&lX[nxt][row][c4] = p;
      }
      if (wave == 0) __builtin_amdgcn_s_wait_tensorcnt(0);
    }
    __syncthreads();
  }

  // ---- LoRA epilogue: one K=32 WMMA per tile (upper K half is zeros) ----
  const int batch = mBlk >> 11;
  {
    AFrag ta[4], bb[2];
    #pragma unroll
    for (int i = 0; i < 4; ++i) {
      const unsigned short* p =
          t2 + (size_t)(mBlk + wm * 64 + i * 16 + l15) * 32 + hi * 8;
      ta[i].q[0] = *(const uint4*)p;
      ta[i].q[1] = *(const uint4*)(p + 16);
    }
    #pragma unroll
    for (int j = 0; j < 2; ++j) {
      const unsigned short* p =
          Beps + ((size_t)batch * D + nBlk + wn * 32 + j * 16 + l15) * 32 + hi * 8;
      bb[j].q[0] = *(const uint4*)p;
      bb[j].q[1] = *(const uint4*)(p + 16);
    }
    #pragma unroll
    for (int i = 0; i < 4; ++i)
      #pragma unroll
      for (int j = 0; j < 2; ++j)
        acc[i][j].v = __builtin_amdgcn_wmma_f32_16x16x32_bf16(
            false, ta[i].v, false, bb[j].v, (short)0, acc[i][j].v, false, false);
  }

  // ---- bias + store (C layout: VGPR v -> M = v + 8*hi, N = l15) ----
  float bv[2];
  #pragma unroll
  for (int j = 0; j < 2; ++j) bv[j] = bias[nBlk + wn * 32 + j * 16 + l15];

  #pragma unroll
  for (int i = 0; i < 4; ++i) {
    #pragma unroll
    for (int v = 0; v < 8; ++v) {
      const size_t row = (size_t)(mBlk + wm * 64 + i * 16 + v + 8 * hi) * D;
      #pragma unroll
      for (int j = 0; j < 2; ++j)
        out[row + nBlk + wn * 32 + j * 16 + l15] = acc[i][j].f[v] + bv[j];
    }
  }
}

// ---------------- launcher ----------------
extern "C" void kernel_launch(void* const* d_in, const int* in_sizes, int n_in,
                              void* d_out, int out_size, void* d_ws, size_t ws_size,
                              hipStream_t stream) {
  const float* x = (const float*)d_in[0];
  const float* W = (const float*)d_in[1];
  const float* b = (const float*)d_in[2];
  const float* A = (const float*)d_in[3];
  const float* B = (const float*)d_in[4];
  float* out = (float*)d_out;

  char* ws = (char*)d_ws;
  // ws layout (bytes): Wb 32MB | A1 1MB | Beps 2MB | t2 1MB  (36 MB total)
  unsigned short* Wb   = (unsigned short*)ws;
  unsigned short* A1   = (unsigned short*)(ws + 33554432);
  unsigned short* Beps = (unsigned short*)(ws + 33554432 + 1048576);
  unsigned short* t2   = (unsigned short*)(ws + 33554432 + 1048576 + 2097152);

  const int n2 = D * D / 2;  // float2 chunks of W
  convert_w_bf16<<<(n2 + 255) / 256, 256, 0, stream>>>(W, (unsigned int*)Wb, n2);
  build_A1<<<(BS * RK * D + 255) / 256, 256, 0, stream>>>(A, A1);
  build_Beps<<<(BS * D * 32 + 255) / 256, 256, 0, stream>>>(B, Beps);
  lora_stage1<<<M_TOT / 128, 256, 0, stream>>>(x, A1, t2);

  dim3 grid(D / 128, M_TOT / 128);  // (32, 128)
  lora_gemm<<<grid, 256, 0, stream>>>(x, Wb, b, t2, Beps, out);
}
